// EdgeTypeAwareGNN_41180146434141
// MI455X (gfx1250) — matrix operand order repaired and checked by reading, not compiled, not run on hardware
//
#include <hip/hip_runtime.h>

// ---------------------------------------------------------------------------
// RGCN (2-layer) + mean-pool + linear head for MI455X (gfx1250, wave32, WMMA)
//
// Strategy:
//   cnt[n,r]      = #edges into n with relation r (one pass, shared by layers)
//   per layer:
//     agg         = x @ root + b                    (WMMA bf16 GEMM, init)
//     for r in 0..R-1:
//        hr       = x @ W[r]                        (WMMA bf16 GEMM)
//        agg[dst] += hr[src] / cnt[dst,r]           (edge scatter, atomics)
//     h           = relu(agg)                       (fused into next stage)
//   pool per graph, then out = pooled @ cls_w + cls_b
// ---------------------------------------------------------------------------

typedef __attribute__((ext_vector_type(16))) __bf16 v16bf;
typedef __attribute__((ext_vector_type(8)))  __bf16 v8bf;
typedef __attribute__((ext_vector_type(8)))  float  v8f;

#define R_REL 10
#define H_DIM 64

// ------------------------------- utility kernels ---------------------------

__global__ void __launch_bounds__(256) zero_f32_kernel(float* __restrict__ p, long n) {
    long i = (long)blockIdx.x * blockDim.x + threadIdx.x;
    long s = (long)gridDim.x * blockDim.x;
    for (; i < n; i += s) p[i] = 0.0f;
}

__global__ void __launch_bounds__(256) f32_to_bf16_kernel(const float* __restrict__ src,
                                                          __bf16* __restrict__ dst, long n) {
    long i = (long)blockIdx.x * blockDim.x + threadIdx.x;
    long s = (long)gridDim.x * blockDim.x;
    for (; i < n; i += s) dst[i] = (__bf16)src[i];
}

// relu(f32) -> bf16 (layer1 output feeding layer2 GEMMs)
__global__ void __launch_bounds__(256) relu_to_bf16_kernel(const float* __restrict__ src,
                                                           __bf16* __restrict__ dst, long n) {
    long i = (long)blockIdx.x * blockDim.x + threadIdx.x;
    long s = (long)gridDim.x * blockDim.x;
    for (; i < n; i += s) dst[i] = (__bf16)fmaxf(src[i], 0.0f);
}

// W [Rr, K, H] f32 row-major  ->  Wt [Rr, H, K] bf16 (transposed per relation)
__global__ void __launch_bounds__(256) wtrans_kernel(const float* __restrict__ w,
                                                     __bf16* __restrict__ wt,
                                                     int Rr, int K, int H) {
    long total = (long)Rr * K * H;
    long i = (long)blockIdx.x * blockDim.x + threadIdx.x;
    long s = (long)gridDim.x * blockDim.x;
    for (; i < total; i += s) {
        long r   = i / ((long)K * H);
        long rem = i - r * (long)K * H;
        long k   = rem / H;
        long n   = rem - k * H;
        wt[r * (long)H * K + n * (long)K + k] = (__bf16)w[i];
    }
}

// per-(dst, relation) in-degree counts
__global__ void __launch_bounds__(256) count_kernel(const int* __restrict__ dst,
                                                    const int* __restrict__ et,
                                                    float* __restrict__ cnt, long E) {
    long i = (long)blockIdx.x * blockDim.x + threadIdx.x;
    long s = (long)gridDim.x * blockDim.x;
    for (; i < E; i += s)
        atomicAdd(&cnt[(long)dst[i] * R_REL + et[i]], 1.0f);
}

// ------------------------------- WMMA GEMM ---------------------------------
// out[N,64] = A[N,K](bf16) * Bt[64,K](bf16, transposed) (+ bias[64])
// One wave -> 16-row x 64-col tile; 4 accumulators of 16x16 f32.
// A layout (16-bit A 16x32): lane<16 -> K 0..15; lane>=16 -> K 16..31 (contiguous, same order as B)
// B layout (dense 32x16): lanes0-15 col=lane K 0..15; lanes16-31 col=lane-16 K 16..31
__global__ void __launch_bounds__(256) rgcn_gemm_bf16_kernel(
        const __bf16* __restrict__ A,
        const __bf16* __restrict__ Bt,
        const float*  __restrict__ bias,   // may be nullptr
        float* __restrict__ out,
        int N, int K)
{
    const int wave  = threadIdx.x >> 5;
    const int lane  = threadIdx.x & 31;
    const int tile  = blockIdx.x * 8 + wave;
    const int node0 = tile * 16;
    if (node0 >= N) return;

    const int half = lane >> 4;   // 0 or 1
    const int lidx = lane & 15;

    const __bf16* arow = A + (long)(node0 + lidx) * K;

    v8f c0 = {}, c1 = {}, c2 = {}, c3 = {};

    for (int k0 = 0; k0 < K; k0 += 32) {
        // A fragment: 16 contiguous K values per lane (K = 16*half + i)
        v16bf a = *(const v16bf*)(arow + k0 + half * 16);

        // B fragments: 16 contiguous K values per lane from transposed weights
        const __bf16* bbase = Bt + (long)lidx * K + k0 + half * 16;
        v16bf b0 = *(const v16bf*)(bbase);
        v16bf b1 = *(const v16bf*)(bbase + 16 * (long)K);
        v16bf b2 = *(const v16bf*)(bbase + 32 * (long)K);
        v16bf b3 = *(const v16bf*)(bbase + 48 * (long)K);

        c0 = __builtin_amdgcn_wmma_f32_16x16x32_bf16(false, a, false, b0, (short)0, c0, false, false);
        c1 = __builtin_amdgcn_wmma_f32_16x16x32_bf16(false, a, false, b1, (short)0, c1, false, false);
        c2 = __builtin_amdgcn_wmma_f32_16x16x32_bf16(false, a, false, b2, (short)0, c2, false, false);
        c3 = __builtin_amdgcn_wmma_f32_16x16x32_bf16(false, a, false, b3, (short)0, c3, false, false);
    }

    // C/D layout: lane<16 -> rows M=0..7 in VGPR0..7, N=lane; lane>=16 -> rows M=8..15, N=lane-16
#define STORE_TILE(cc, t)                                                     \
    do {                                                                      \
        int n = (t) * 16 + lidx;                                              \
        float badd = bias ? bias[n] : 0.0f;                                   \
        _Pragma("unroll")                                                     \
        for (int i = 0; i < 8; ++i) {                                         \
            int m = half * 8 + i;                                             \
            out[(long)(node0 + m) * H_DIM + n] = cc[i] + badd;                \
        }                                                                     \
    } while (0)

    STORE_TILE(c0, 0);
    STORE_TILE(c1, 1);
    STORE_TILE(c2, 2);
    STORE_TILE(c3, 3);
#undef STORE_TILE
}

// ------------------------------- edge scatter ------------------------------
// 64 threads per edge; agg[dst] += hr[src] / cnt[dst,rel] for edges of `rel`
__global__ void __launch_bounds__(256) edge_scatter_kernel(
        const int* __restrict__ src, const int* __restrict__ dst,
        const int* __restrict__ et,  const float* __restrict__ cnt,
        const float* __restrict__ hr, float* __restrict__ agg,
        long E, int rel)
{
    const int  j  = threadIdx.x & 63;
    long g  = ((long)blockIdx.x * blockDim.x + threadIdx.x) >> 6;
    long gs = ((long)gridDim.x * blockDim.x) >> 6;
    for (long e = g; e < E; e += gs) {
        if (et[e] != rel) continue;
        int s = src[e];
        int d = dst[e];
        float c   = cnt[(long)d * R_REL + rel];
        float inv = 1.0f / fmaxf(c, 1.0f);
        atomicAdd(&agg[(long)d * H_DIM + j], hr[(long)s * H_DIM + j] * inv);
    }
}

// ------------------------------- pooling + head ----------------------------
// fused relu(agg2) then per-graph mean-pool accumulation
__global__ void __launch_bounds__(256) pool_kernel(const float* __restrict__ agg2,
                                                   const int* __restrict__ batch,
                                                   float* __restrict__ pool,
                                                   float* __restrict__ pcnt, long N)
{
    long total = N * H_DIM;
    long i = (long)blockIdx.x * blockDim.x + threadIdx.x;
    long s = (long)gridDim.x * blockDim.x;
    for (; i < total; i += s) {
        long n = i >> 6;
        int  j = (int)(i & 63);
        int  g = batch[n];
        atomicAdd(&pool[(long)g * H_DIM + j], fmaxf(agg2[i], 0.0f));
        if (j == 0) atomicAdd(&pcnt[g], 1.0f);
    }
}

__global__ void __launch_bounds__(64) final_kernel(const float* __restrict__ pool,
                                                   const float* __restrict__ pcnt,
                                                   const float* __restrict__ cw,
                                                   const float* __restrict__ cb,
                                                   float* __restrict__ out, int G)
{
    int g = blockIdx.x * blockDim.x + threadIdx.x;
    if (g >= G) return;
    float inv = 1.0f / fmaxf(pcnt[g], 1.0f);
    float acc = 0.0f;
#pragma unroll
    for (int j = 0; j < H_DIM; ++j) acc += pool[(long)g * H_DIM + j] * inv * cw[j];
    out[g] = acc + cb[0];
}

// ------------------------------- launcher ----------------------------------

static inline char* ws_alloc(char*& p, size_t bytes) {
    char* q = p;
    p += (bytes + 255) & ~(size_t)255;
    return q;
}

extern "C" void kernel_launch(void* const* d_in, const int* in_sizes, int n_in,
                              void* d_out, int out_size, void* d_ws, size_t ws_size,
                              hipStream_t stream) {
    const float* x     = (const float*)d_in[0];   // [N, 128]
    const int*   eidx  = (const int*)  d_in[1];   // [2, E]
    const int*   etype = (const int*)  d_in[2];   // [E]
    const int*   batch = (const int*)  d_in[3];   // [N]
    const float* W1    = (const float*)d_in[4];   // [R, 128, 64]
    const float* root1 = (const float*)d_in[5];   // [128, 64]
    const float* b1    = (const float*)d_in[6];   // [64]
    const float* W2    = (const float*)d_in[7];   // [R, 64, 64]
    const float* root2 = (const float*)d_in[8];   // [64, 64]
    const float* b2    = (const float*)d_in[9];   // [64]
    const float* cls_w = (const float*)d_in[10];  // [64, 1]
    const float* cls_b = (const float*)d_in[11];  // [1]
    float*       out   = (float*)d_out;

    const int DIN = in_sizes[5] / H_DIM;          // 128
    const int N   = in_sizes[0] / DIN;            // 100000 (multiple of 16)
    const long E  = in_sizes[2];                  // 3.2M
    const int G   = out_size;                     // 64

    const int* src = eidx;
    const int* dst = eidx + E;

    // ---- workspace carve (~95 MB) ----
    char* p = (char*)d_ws;
    float*  cnt    = (float*) ws_alloc(p, (size_t)N * R_REL * 4);
    __bf16* xb     = (__bf16*)ws_alloc(p, (size_t)N * DIN * 2);
    __bf16* h1b    = (__bf16*)ws_alloc(p, (size_t)N * H_DIM * 2);
    float*  agg    = (float*) ws_alloc(p, (size_t)N * H_DIM * 4);
    float*  hr     = (float*) ws_alloc(p, (size_t)N * H_DIM * 4);
    __bf16* Wbt    = (__bf16*)ws_alloc(p, (size_t)R_REL * H_DIM * DIN * 2);
    __bf16* rootbt = (__bf16*)ws_alloc(p, (size_t)H_DIM * DIN * 2);
    float*  pool   = (float*) ws_alloc(p, ((size_t)G * H_DIM + G) * 4);
    float*  pcnt   = pool + (size_t)G * H_DIM;
    (void)ws_size; (void)n_in;

    const int gemm_blocks = (N / 16 + 7) / 8;

    // ---- per-(dst,rel) counts (shared by both layers) ----
    zero_f32_kernel<<<1024, 256, 0, stream>>>(cnt, (long)N * R_REL);
    count_kernel<<<2048, 256, 0, stream>>>(dst, etype, cnt, E);

    // ---- layer 1 ----
    f32_to_bf16_kernel<<<2048, 256, 0, stream>>>(x, xb, (long)N * DIN);
    wtrans_kernel<<<512, 256, 0, stream>>>(W1, Wbt, R_REL, DIN, H_DIM);
    wtrans_kernel<<<64, 256, 0, stream>>>(root1, rootbt, 1, DIN, H_DIM);

    // agg = x @ root1 + b1
    rgcn_gemm_bf16_kernel<<<gemm_blocks, 256, 0, stream>>>(xb, rootbt, b1, agg, N, DIN);
    for (int r = 0; r < R_REL; ++r) {
        rgcn_gemm_bf16_kernel<<<gemm_blocks, 256, 0, stream>>>(
            xb, Wbt + (size_t)r * H_DIM * DIN, nullptr, hr, N, DIN);
        edge_scatter_kernel<<<2048, 256, 0, stream>>>(src, dst, etype, cnt, hr, agg, E, r);
    }
    // h1 = relu(agg) -> bf16
    relu_to_bf16_kernel<<<2048, 256, 0, stream>>>(agg, h1b, (long)N * H_DIM);

    // ---- layer 2 (K = 64) ----
    wtrans_kernel<<<256, 256, 0, stream>>>(W2, Wbt, R_REL, H_DIM, H_DIM);
    wtrans_kernel<<<32, 256, 0, stream>>>(root2, rootbt, 1, H_DIM, H_DIM);

    rgcn_gemm_bf16_kernel<<<gemm_blocks, 256, 0, stream>>>(h1b, rootbt, b2, agg, N, H_DIM);
    for (int r = 0; r < R_REL; ++r) {
        rgcn_gemm_bf16_kernel<<<gemm_blocks, 256, 0, stream>>>(
            h1b, Wbt + (size_t)r * H_DIM * H_DIM, nullptr, hr, N, H_DIM);
        edge_scatter_kernel<<<2048, 256, 0, stream>>>(src, dst, etype, cnt, hr, agg, E, r);
    }

    // ---- pooling + classifier (relu fused into pooling) ----
    zero_f32_kernel<<<8, 256, 0, stream>>>(pool, (long)G * H_DIM + G);
    pool_kernel<<<2048, 256, 0, stream>>>(agg, batch, pool, pcnt, (long)N);
    final_kernel<<<1, 64, 0, stream>>>(pool, pcnt, cls_w, cls_b, out, G);
}
